// OptimizedSpecializedBranch_87230785782090
// MI455X (gfx1250) — compile-verified, matrix operand order checked by reading, not run on hardware
//
#include <hip/hip_runtime.h>
#include <hip/hip_bf16.h>
#include <math.h>

// ---------------------------------------------------------------------------
// Transformer encoder layer for MI455X (gfx1250), bf16 WMMA w/ fp32 accum.
// B=2, S=2048, D=1024, H=32, HD=32, DFF=4096.
// GEMMs: double-buffered LDS fed by the gfx1250 async global->LDS engine
// (ASYNCcnt) so tile fill overlaps WMMA issue; sync fallback if the builtin
// is unavailable.
// ---------------------------------------------------------------------------

typedef __attribute__((ext_vector_type(16))) __bf16 v16bf;
typedef __attribute__((ext_vector_type(8)))  float  v8f;

#define DMODEL 1024
#define NHEAD  32
#define HDIM   32
#define DFFN   4096
#define SEQ    2048
#define NBATCH 2
#define MTOK   (NBATCH * SEQ)   // 4096 rows

#if defined(__has_builtin)
#if __has_builtin(__builtin_amdgcn_global_load_async_to_lds_b128)
#define HAS_ASYNC_LDS 1
#endif
#endif

#ifdef HAS_ASYNC_LDS
// Builtin signature (from hipcc diagnostics): param0 is
// 'int __attribute__((vector_size(16))) __device__ *'  (addrspace(1)),
// i.e. a GCC-vector-of-4-int pointer in global memory. Launder the flat
// pointers through integers: flat->global is numerically identity, flat->LDS
// is a low-32-bit truncate.
typedef int v4i_ __attribute__((vector_size(16)));
#define ASYNC_CP128(gsrc, ldst)                                               \
  __builtin_amdgcn_global_load_async_to_lds_b128(                             \
      (__attribute__((address_space(1))) v4i_*)(uintptr_t)(gsrc),             \
      (__attribute__((address_space(3))) v4i_*)(uint32_t)(uintptr_t)(ldst),   \
      0, 0)

__device__ inline void wait_async0() {
#if __has_builtin(__builtin_amdgcn_s_wait_asynccnt)
  __builtin_amdgcn_s_wait_asynccnt(0);
#else
  asm volatile("s_wait_asynccnt 0x0" ::: "memory");
#endif
}
#endif

// ---------------------------------------------------------------------------
// fp32 -> bf16 cast (weights / staging). n must be a multiple of 4.
// ---------------------------------------------------------------------------
__global__ void cast_f32_to_bf16_k(const float* __restrict__ src,
                                   __bf16* __restrict__ dst, int n) {
  int i = (blockIdx.x * blockDim.x + threadIdx.x) * 4;
  if (i + 3 < n) {
    float4 f = *(const float4*)(src + i);
    dst[i + 0] = (__bf16)f.x;
    dst[i + 1] = (__bf16)f.y;
    dst[i + 2] = (__bf16)f.z;
    dst[i + 3] = (__bf16)f.w;
  }
}

// ---------------------------------------------------------------------------
// Fused (a [+ b]) -> LayerNorm -> {fp32 out, bf16 out}. One 256-thread block
// per row of D=1024. Grid = MTOK.
// ---------------------------------------------------------------------------
__global__ void __launch_bounds__(256) add_layernorm_k(
    const float* __restrict__ a, const float* __restrict__ badd,
    const float* __restrict__ g, const float* __restrict__ be,
    float* __restrict__ outF, __bf16* __restrict__ outB) {
  __shared__ float red1[256];
  __shared__ float red2[256];
  const int row = blockIdx.x;
  const float* ap = a + (size_t)row * DMODEL;
  const float* bp = badd ? badd + (size_t)row * DMODEL : nullptr;

  float v[4];
  float s = 0.f, ss = 0.f;
#pragma unroll
  for (int i = 0; i < 4; ++i) {
    int c = threadIdx.x + i * 256;
    float x = ap[c];
    if (bp) x += bp[c];
    v[i] = x;
    s += x;
    ss += x * x;
  }
  red1[threadIdx.x] = s;
  red2[threadIdx.x] = ss;
  __syncthreads();
  for (int off = 128; off > 0; off >>= 1) {
    if (threadIdx.x < off) {
      red1[threadIdx.x] += red1[threadIdx.x + off];
      red2[threadIdx.x] += red2[threadIdx.x + off];
    }
    __syncthreads();
  }
  const float mean = red1[0] * (1.0f / DMODEL);
  const float var  = red2[0] * (1.0f / DMODEL) - mean * mean;
  const float rstd = rsqrtf(var + 1e-5f);
#pragma unroll
  for (int i = 0; i < 4; ++i) {
    int c = threadIdx.x + i * 256;
    float y = (v[i] - mean) * rstd * g[c] + be[c];
    size_t idx = (size_t)row * DMODEL + c;
    if (outF) outF[idx] = y;
    if (outB) outB[idx] = (__bf16)y;
  }
}

// ---------------------------------------------------------------------------
// Tiled WMMA GEMM: C[M,N] = A[M,K](bf16) @ W[N,K]^T(bf16) + bias, optional
// exact GELU, writes fp32 and/or bf16. Block tile 128x128, BK=32, 8 waves,
// wave grid 4(M) x 2(N), per-wave 32x64 (2x4 WMMA 16x16 accumulators).
// Double-buffered LDS; async global->LDS fill overlaps WMMA issue.
// M,N multiples of 128; K multiple of 32.
// ---------------------------------------------------------------------------
__global__ void __launch_bounds__(256) gemm_bf16_wmma_k(
    const __bf16* __restrict__ A, const __bf16* __restrict__ W,
    const float* __restrict__ bias, float* __restrict__ outF,
    __bf16* __restrict__ outB, int Mdim, int Ndim, int Kdim, int act) {
  const int tid  = threadIdx.x;
  const int lane = tid & 31;
  const int wave = tid >> 5;
  const int wm   = wave >> 1;  // 0..3
  const int wn   = wave & 1;   // 0..1
  const int m0   = blockIdx.y * 128;
  const int n0   = blockIdx.x * 128;
  const int ln   = lane & 15;
  const int kk   = (lane >> 4) * 16;
  const int rhalf = (lane >> 4) * 8;

  v8f zero = {};
  v8f acc[2][4];
#pragma unroll
  for (int i = 0; i < 2; ++i)
#pragma unroll
    for (int j = 0; j < 4; ++j) acc[i][j] = zero;

  // Per-thread staging coordinates: 2 groups of 8 bf16 (16B) per tile each
  // for A and W.
  const int e0 = tid;            // group ids tid, tid+256 of 512
  const int r0 = e0 >> 2, c0 = (e0 & 3) * 8;
  const int e1 = tid + 256;
  const int r1 = e1 >> 2, c1 = (e1 & 3) * 8;

#ifdef HAS_ASYNC_LDS
  __shared__ __align__(32) __bf16 As[2][128 * 32];
  __shared__ __align__(32) __bf16 Bs[2][128 * 32];

  auto stage_async = [&](int buf, int k0) {
    ASYNC_CP128(&A[(size_t)(m0 + r0) * Kdim + k0 + c0], &As[buf][r0 * 32 + c0]);
    ASYNC_CP128(&W[(size_t)(n0 + r0) * Kdim + k0 + c0], &Bs[buf][r0 * 32 + c0]);
    ASYNC_CP128(&A[(size_t)(m0 + r1) * Kdim + k0 + c1], &As[buf][r1 * 32 + c1]);
    ASYNC_CP128(&W[(size_t)(n0 + r1) * Kdim + k0 + c1], &Bs[buf][r1 * 32 + c1]);
  };

  stage_async(0, 0);
  wait_async0();
  __syncthreads();

  int cur = 0;
  for (int k0 = 0; k0 < Kdim; k0 += 32) {
    const bool more = (k0 + 32) < Kdim;
    // Kick off the next tile fill through the async LDS engine, then compute
    // on the current buffer while it streams in.
    if (more) stage_async(cur ^ 1, k0 + 32);

    v16bf afrag[2], bfrag[4];
#pragma unroll
    for (int sm = 0; sm < 2; ++sm)
      afrag[sm] = *(const v16bf*)&As[cur][(wm * 32 + sm * 16 + ln) * 32 + kk];
#pragma unroll
    for (int sn = 0; sn < 4; ++sn)
      bfrag[sn] = *(const v16bf*)&Bs[cur][(wn * 64 + sn * 16 + ln) * 32 + kk];

#pragma unroll
    for (int sm = 0; sm < 2; ++sm)
#pragma unroll
      for (int sn = 0; sn < 4; ++sn)
        acc[sm][sn] = __builtin_amdgcn_wmma_f32_16x16x32_bf16(
            false, afrag[sm], false, bfrag[sn], (short)0, acc[sm][sn], false,
            false);

    if (more) {
      wait_async0();     // this wave's fills for buffer cur^1 are in LDS
      __syncthreads();   // everyone's fills are in LDS
    }
    cur ^= 1;
  }
#else
  __shared__ __align__(32) __bf16 As[128 * 32];
  __shared__ __align__(32) __bf16 Bs[128 * 32];

  for (int k0 = 0; k0 < Kdim; k0 += 32) {
    *(uint4*)&As[r0 * 32 + c0] =
        *(const uint4*)&A[(size_t)(m0 + r0) * Kdim + k0 + c0];
    *(uint4*)&Bs[r0 * 32 + c0] =
        *(const uint4*)&W[(size_t)(n0 + r0) * Kdim + k0 + c0];
    *(uint4*)&As[r1 * 32 + c1] =
        *(const uint4*)&A[(size_t)(m0 + r1) * Kdim + k0 + c1];
    *(uint4*)&Bs[r1 * 32 + c1] =
        *(const uint4*)&W[(size_t)(n0 + r1) * Kdim + k0 + c1];
    __syncthreads();

    if (k0 + 32 < Kdim) {
      __builtin_prefetch(&A[(size_t)(m0 + (tid >> 1)) * Kdim + k0 + 32], 0, 1);
      __builtin_prefetch(&W[(size_t)(n0 + (tid >> 1)) * Kdim + k0 + 32], 0, 1);
    }

    v16bf afrag[2], bfrag[4];
#pragma unroll
    for (int sm = 0; sm < 2; ++sm)
      afrag[sm] = *(const v16bf*)&As[(wm * 32 + sm * 16 + ln) * 32 + kk];
#pragma unroll
    for (int sn = 0; sn < 4; ++sn)
      bfrag[sn] = *(const v16bf*)&Bs[(wn * 64 + sn * 16 + ln) * 32 + kk];

#pragma unroll
    for (int sm = 0; sm < 2; ++sm)
#pragma unroll
      for (int sn = 0; sn < 4; ++sn)
        acc[sm][sn] = __builtin_amdgcn_wmma_f32_16x16x32_bf16(
            false, afrag[sm], false, bfrag[sn], (short)0, acc[sm][sn], false,
            false);
    __syncthreads();
  }
#endif

  // Epilogue: bias, optional exact GELU, store fp32/bf16.
#pragma unroll
  for (int sm = 0; sm < 2; ++sm) {
#pragma unroll
    for (int sn = 0; sn < 4; ++sn) {
      int ccol = n0 + wn * 64 + sn * 16 + ln;
      float bb = bias[ccol];
#pragma unroll
      for (int j = 0; j < 8; ++j) {
        int rrow = m0 + wm * 32 + sm * 16 + rhalf + j;
        float vo = acc[sm][sn][j] + bb;
        if (act == 1) vo = 0.5f * vo * (1.0f + erff(vo * 0.70710678118f));
        size_t idx = (size_t)rrow * Ndim + ccol;
        if (outF) outF[idx] = vo;
        if (outB) outB[idx] = (__bf16)vo;
      }
    }
  }
}

// ---------------------------------------------------------------------------
// Flash attention, one WG per (b, h, 128-query chunk); 8 waves, each wave owns
// a 16-query tile. HD=32 -> Q fragment is exactly one WMMA A operand.
// Online softmax over 32-key chunks: 2 score WMMAs + 2 context WMMAs/chunk.
// ---------------------------------------------------------------------------
__global__ void __launch_bounds__(256) attention_wmma_k(
    const __bf16* __restrict__ Q, const __bf16* __restrict__ Kmat,
    const __bf16* __restrict__ V, __bf16* __restrict__ O) {
  __shared__ __align__(32) __bf16 Vt[32 * 32];        // [hd][key], per chunk
  __shared__ __align__(32) __bf16 Pbuf[8][16 * 32];   // wave-private P tile

  const int tid  = threadIdx.x;
  const int lane = tid & 31;
  const int wave = tid >> 5;
  const int h = blockIdx.y;
  const int b = blockIdx.z;
  const int q0 = blockIdx.x * 128 + wave * 16;  // seq index of wave's q tile
  const size_t baseTok = (size_t)b * SEQ;
  const int ln = lane & 15;
  const int kk = (lane >> 4) * 16;
  const int rhalf = (lane >> 4) * 8;
  const float scale = 0.17677669529663687f;  // 1/sqrt(HD)

  // Q fragment (16 x 32), one v16bf per lane.
  v16bf qf =
      *(const v16bf*)&Q[(baseTok + q0 + ln) * DMODEL + h * HDIM + kk];

  v8f ctx0 = {}, ctx1 = {};
  float mrow[8], drow[8];
#pragma unroll
  for (int j = 0; j < 8; ++j) {
    mrow[j] = -1e30f;
    drow[j] = 0.0f;
  }

  for (int kc = 0; kc < SEQ / 32; ++kc) {
    const int key0 = kc * 32;

    // Stage V chunk transposed: Vt[hd][key]; 32 keys x 32 hd.
    {
      int key = tid >> 3;          // 0..31
      int hd4 = (tid & 7) * 4;     // 0,4,...,28
      const __bf16* vp = &V[(baseTok + key0 + key) * DMODEL + h * HDIM + hd4];
      __bf16 t0 = vp[0], t1 = vp[1], t2 = vp[2], t3 = vp[3];
      Vt[(hd4 + 0) * 32 + key] = t0;
      Vt[(hd4 + 1) * 32 + key] = t1;
      Vt[(hd4 + 2) * 32 + key] = t2;
      Vt[(hd4 + 3) * 32 + key] = t3;
    }
    __syncthreads();

    // Scores: 16 queries x 32 keys via two WMMAs (K frags straight from L2).
    v8f sc[2];
#pragma unroll
    for (int kt = 0; kt < 2; ++kt) {
      v16bf kf = *(const v16bf*)&Kmat[(baseTok + key0 + kt * 16 + ln) * DMODEL +
                                      h * HDIM + kk];
      v8f z = {};
      sc[kt] = __builtin_amdgcn_wmma_f32_16x16x32_bf16(false, qf, false, kf,
                                                       (short)0, z, false,
                                                       false);
    }

    // Online softmax update per row (row = rhalf + j, cols across 16 lanes).
#pragma unroll
    for (int j = 0; j < 8; ++j) {
      float s0 = sc[0][j] * scale;
      float s1 = sc[1][j] * scale;
      float tmax = fmaxf(s0, s1);
#pragma unroll
      for (int msk = 1; msk < 16; msk <<= 1)
        tmax = fmaxf(tmax, __shfl_xor(tmax, msk, 32));
      float mnew = fmaxf(mrow[j], tmax);
      float corr = __expf(mrow[j] - mnew);
      mrow[j] = mnew;
      float p0 = __expf(s0 - mnew);
      float p1 = __expf(s1 - mnew);
      float psum = p0 + p1;
#pragma unroll
      for (int msk = 1; msk < 16; msk <<= 1)
        psum += __shfl_xor(psum, msk, 32);
      drow[j] = drow[j] * corr + psum;
      ctx0[j] = ctx0[j] * corr;
      ctx1[j] = ctx1[j] * corr;
      int r = rhalf + j;
      Pbuf[wave][r * 32 + ln]      = (__bf16)p0;
      Pbuf[wave][r * 32 + 16 + ln] = (__bf16)p1;
    }

    // Context accumulate: P(16x32) @ Vchunk(32x32) as two 16-col WMMAs.
    v16bf pa = *(const v16bf*)&Pbuf[wave][ln * 32 + kk];
    v16bf v0 = *(const v16bf*)&Vt[ln * 32 + kk];
    v16bf v1 = *(const v16bf*)&Vt[(16 + ln) * 32 + kk];
    ctx0 = __builtin_amdgcn_wmma_f32_16x16x32_bf16(false, pa, false, v0,
                                                   (short)0, ctx0, false,
                                                   false);
    ctx1 = __builtin_amdgcn_wmma_f32_16x16x32_bf16(false, pa, false, v1,
                                                   (short)0, ctx1, false,
                                                   false);
    __syncthreads();  // before Vt is overwritten
  }

  // Normalize and write context (bf16, [B,S,D] with head slice contiguous).
#pragma unroll
  for (int j = 0; j < 8; ++j) {
    float inv = 1.0f / drow[j];
    int r = rhalf + j;
    size_t o = (baseTok + q0 + r) * DMODEL + h * HDIM;
    O[o + ln]      = (__bf16)(ctx0[j] * inv);
    O[o + 16 + ln] = (__bf16)(ctx1[j] * inv);
  }
}

// ---------------------------------------------------------------------------
// Host-side orchestration.
// ---------------------------------------------------------------------------
extern "C" void kernel_launch(void* const* d_in, const int* in_sizes, int n_in,
                              void* d_out, int out_size, void* d_ws,
                              size_t ws_size, hipStream_t stream) {
  const float* x       = (const float*)d_in[0];
  const float* ln_in_g = (const float*)d_in[1];
  const float* ln_in_b = (const float*)d_in[2];
  const float* wq = (const float*)d_in[3];
  const float* bq = (const float*)d_in[4];
  const float* wk = (const float*)d_in[5];
  const float* bk = (const float*)d_in[6];
  const float* wv = (const float*)d_in[7];
  const float* bv = (const float*)d_in[8];
  const float* wo = (const float*)d_in[9];
  const float* bo = (const float*)d_in[10];
  const float* ln1_g = (const float*)d_in[11];
  const float* ln1_b = (const float*)d_in[12];
  const float* w1 = (const float*)d_in[13];
  const float* b1 = (const float*)d_in[14];
  const float* w2 = (const float*)d_in[15];
  const float* b2 = (const float*)d_in[16];
  const float* ln2_g = (const float*)d_in[17];
  const float* ln2_b = (const float*)d_in[18];
  float* out = (float*)d_out;

  char* ws = (char*)d_ws;
  size_t off = 0;
  auto alloc = [&](size_t bytes) -> char* {
    char* p = ws + off;
    off += (bytes + 255) & ~(size_t)255;
    return p;
  };

  __bf16* wqb = (__bf16*)alloc((size_t)DMODEL * DMODEL * 2);
  __bf16* wkb = (__bf16*)alloc((size_t)DMODEL * DMODEL * 2);
  __bf16* wvb = (__bf16*)alloc((size_t)DMODEL * DMODEL * 2);
  __bf16* wob = (__bf16*)alloc((size_t)DMODEL * DMODEL * 2);
  __bf16* w1b = (__bf16*)alloc((size_t)DFFN * DMODEL * 2);
  __bf16* w2b = (__bf16*)alloc((size_t)DMODEL * DFFN * 2);
  __bf16* nxb = (__bf16*)alloc((size_t)MTOK * DMODEL * 2);
  __bf16* qb  = (__bf16*)alloc((size_t)MTOK * DMODEL * 2);
  __bf16* kb  = (__bf16*)alloc((size_t)MTOK * DMODEL * 2);
  __bf16* vb  = (__bf16*)alloc((size_t)MTOK * DMODEL * 2);
  __bf16* ctxb = (__bf16*)alloc((size_t)MTOK * DMODEL * 2);
  float*  attnF = (float*)alloc((size_t)MTOK * DMODEL * 4);
  float*  x1F   = (float*)alloc((size_t)MTOK * DMODEL * 4);
  __bf16* x1b   = (__bf16*)alloc((size_t)MTOK * DMODEL * 2);
  __bf16* ffhb  = (__bf16*)alloc((size_t)MTOK * DFFN * 2);
  float*  ffF   = (float*)alloc((size_t)MTOK * DMODEL * 4);

  auto cast = [&](const float* s, __bf16* d, size_t n) {
    cast_f32_to_bf16_k<<<(unsigned)(n / 1024), 256, 0, stream>>>(s, d, (int)n);
  };

  // 1) weights -> bf16
  cast(wq, wqb, (size_t)DMODEL * DMODEL);
  cast(wk, wkb, (size_t)DMODEL * DMODEL);
  cast(wv, wvb, (size_t)DMODEL * DMODEL);
  cast(wo, wob, (size_t)DMODEL * DMODEL);
  cast(w1, w1b, (size_t)DFFN * DMODEL);
  cast(w2, w2b, (size_t)DMODEL * DFFN);

  // 2) input LayerNorm -> bf16 activations
  add_layernorm_k<<<MTOK, 256, 0, stream>>>(x, nullptr, ln_in_g, ln_in_b,
                                            nullptr, nxb);

  // 3) Q/K/V projections (M=4096, N=1024, K=1024)
  dim3 gProj(DMODEL / 128, MTOK / 128);
  gemm_bf16_wmma_k<<<gProj, 256, 0, stream>>>(nxb, wqb, bq, nullptr, qb, MTOK,
                                              DMODEL, DMODEL, 0);
  gemm_bf16_wmma_k<<<gProj, 256, 0, stream>>>(nxb, wkb, bk, nullptr, kb, MTOK,
                                              DMODEL, DMODEL, 0);
  gemm_bf16_wmma_k<<<gProj, 256, 0, stream>>>(nxb, wvb, bv, nullptr, vb, MTOK,
                                              DMODEL, DMODEL, 0);

  // 4) attention
  dim3 gAttn(SEQ / 128, NHEAD, NBATCH);
  attention_wmma_k<<<gAttn, 256, 0, stream>>>(qb, kb, vb, ctxb);

  // 5) output projection -> fp32 attn_out
  gemm_bf16_wmma_k<<<gProj, 256, 0, stream>>>(ctxb, wob, bo, attnF, nullptr,
                                              MTOK, DMODEL, DMODEL, 0);

  // 6) x1 = LN1(x + attn_out) -> fp32 (residual base) + bf16 (FFN input)
  add_layernorm_k<<<MTOK, 256, 0, stream>>>(x, attnF, ln1_g, ln1_b, x1F, x1b);

  // 7) FFN1 + exact GELU (M=4096, N=4096, K=1024)
  dim3 gFf1(DFFN / 128, MTOK / 128);
  gemm_bf16_wmma_k<<<gFf1, 256, 0, stream>>>(x1b, w1b, b1, nullptr, ffhb, MTOK,
                                             DFFN, DMODEL, 1);

  // 8) FFN2 (M=4096, N=1024, K=4096) -> fp32
  gemm_bf16_wmma_k<<<gProj, 256, 0, stream>>>(ffhb, w2b, b2, ffF, nullptr,
                                              MTOK, DMODEL, DFFN, 0);

  // 9) out = LN2(x1 + ff)
  add_layernorm_k<<<MTOK, 256, 0, stream>>>(x1F, ffF, ln2_g, ln2_b, out,
                                            nullptr);
}
